// QuaternionAttention_7189775253859
// MI455X (gfx1250) — matrix operand order repaired and checked by reading, not compiled, not run on hardware
//
#include <hip/hip_runtime.h>

typedef __attribute__((ext_vector_type(16))) _Float16 v16h;
typedef __attribute__((ext_vector_type(8)))  _Float16 v8h;
typedef __attribute__((ext_vector_type(4)))  _Float16 v4h;
typedef __attribute__((ext_vector_type(8)))  float    v8f;

// Problem dims (fixed by the reference)
#define BB   4
#define SS   2048
#define DD   1024
#define HH   8
#define HDIM 128
#define MTOT (BB * SS)   // 8192

static __device__ __forceinline__ v8f wmma16x16x32(v16h a, v16h b, v8f c) {
    // D = A(16x32 f16) * B(32x16 f16) + C(16x16 f32)
    return __builtin_amdgcn_wmma_f32_16x16x32_f16(
        /*neg_a=*/false, a, /*neg_b=*/false, b,
        /*c_mod=*/(short)0, c, /*reuse_a=*/false, /*reuse_b=*/false);
}

// ---------------------------------------------------------------------------
// GEMM with bias: C[M,N] = A[M,K](f32) @ W[K,N](f32) + bias, f16 WMMA, f32 acc
// Block tile 128x128, K-chunk 64, 8 waves; each wave computes 32x64.
// ---------------------------------------------------------------------------
constexpr int BM = 128, BN = 128, BK = 64;
constexpr int LDA_S = BK + 8;   // 72 halves = 144B rows (multiple of 16B)
constexpr int LDB_S = BK + 8;

template <bool WRITE_HALF>
__global__ __launch_bounds__(256)
void gemm_bias_kernel(const float* __restrict__ A, const float* __restrict__ W,
                      const float* __restrict__ bias,
                      float* __restrict__ Cf, _Float16* __restrict__ Ch,
                      int M, int N, int K)
{
    __shared__ _Float16 As[BM * LDA_S];          // [m][k]
    __shared__ _Float16 Bs[BN * LDB_S];          // [n][k]  (W tile transposed)

    const int tid  = threadIdx.x;
    const int lane = tid & 31;
    const int wid  = tid >> 5;
    const int wm   = (wid & 3) * 32;             // wave row base in tile
    const int wn   = (wid >> 2) * 64;            // wave col base in tile
    const int m0   = blockIdx.y * BM;
    const int n0   = blockIdx.x * BN;

    const v8f vz = {0.f,0.f,0.f,0.f,0.f,0.f,0.f,0.f};
    v8f acc[2][4];
    #pragma unroll
    for (int i = 0; i < 2; ++i)
        #pragma unroll
        for (int j = 0; j < 4; ++j) acc[i][j] = vz;

    for (int kc = 0; kc < K; kc += BK) {
        __syncthreads();
        // Stage A tile 128x64: f32 global -> f16 LDS (coalesced float4 loads)
        #pragma unroll
        for (int i = 0; i < 8; ++i) {
            int idx = tid + i * 256;             // 0..2047
            int r = idx >> 4, c4 = idx & 15;
            float4 f = *(const float4*)(A + (size_t)(m0 + r) * K + kc + c4 * 4);
            v4h hv = { (_Float16)f.x, (_Float16)f.y, (_Float16)f.z, (_Float16)f.w };
            *(v4h*)(As + r * LDA_S + c4 * 4) = hv;
        }
        // Stage W tile 64x128 transposed into Bs[n][k]
        #pragma unroll
        for (int i = 0; i < 8; ++i) {
            int idx = tid + i * 256;
            int kr = idx >> 5, n4 = idx & 31;
            float4 f = *(const float4*)(W + (size_t)(kc + kr) * N + n0 + n4 * 4);
            Bs[(n4 * 4 + 0) * LDB_S + kr] = (_Float16)f.x;
            Bs[(n4 * 4 + 1) * LDB_S + kr] = (_Float16)f.y;
            Bs[(n4 * 4 + 2) * LDB_S + kr] = (_Float16)f.z;
            Bs[(n4 * 4 + 3) * LDB_S + kr] = (_Float16)f.w;
        }
        // Prefetch next K-chunk into cache (global_prefetch_b8)
        if (kc + BK < K) {
            __builtin_prefetch(A + (size_t)(m0 + (tid >> 1)) * K + kc + BK, 0, 3);
            __builtin_prefetch(W + (size_t)(kc + BK + (tid >> 2)) * N + n0, 0, 3);
        }
        __syncthreads();

        #pragma unroll
        for (int kk = 0; kk < BK; kk += 32) {
            // A fragments for the wave's two 16-row tiles
            v16h af[2];
            #pragma unroll
            for (int mt = 0; mt < 2; ++mt) {
                int row = wm + mt * 16 + (lane & 15);
                int kb  = kk + (lane >> 4) * 8;
                union { v16h v; v8h h[2]; } u;
                u.h[0] = *(const v8h*)(As + row * LDA_S + kb);
                u.h[1] = *(const v8h*)(As + row * LDA_S + kb + 16);
                af[mt] = u.v;
            }
            #pragma unroll
            for (int nt = 0; nt < 4; ++nt) {
                int col = wn + nt * 16 + (lane & 15);
                int kb  = kk + (lane >> 4) * 8;
                union { v16h v; v8h h[2]; } u;
                u.h[0] = *(const v8h*)(Bs + col * LDB_S + kb);
                u.h[1] = *(const v8h*)(Bs + col * LDB_S + kb + 16);
                acc[0][nt] = wmma16x16x32(af[0], u.v, acc[0][nt]);
                acc[1][nt] = wmma16x16x32(af[1], u.v, acc[1][nt]);
            }
        }
    }

    // Epilogue: C/D layout -> memory (reg r: M = r + (lane>>4)*8, N = lane&15)
    #pragma unroll
    for (int mt = 0; mt < 2; ++mt) {
        #pragma unroll
        for (int nt = 0; nt < 4; ++nt) {
            int col = n0 + wn + nt * 16 + (lane & 15);
            float bv = bias[col];
            #pragma unroll
            for (int r = 0; r < 8; ++r) {
                int row = m0 + wm + mt * 16 + r + ((lane >> 4) * 8);
                float v = acc[mt][nt][r] + bv;
                if (WRITE_HALF) Ch[(size_t)row * N + col] = (_Float16)v;
                else            Cf[(size_t)row * N + col] = v;
            }
        }
    }
}

// ---------------------------------------------------------------------------
// Flash attention: one block per (b, h, 128-row Q tile); 8 waves x 16 q-rows.
// KV tiles of 128; online softmax in f32; QK^T and P@V via WMMA f16.
// ---------------------------------------------------------------------------
constexpr int LQS = HDIM + 8;    // Qs [q][d]
constexpr int LKS = HDIM + 8;    // Ks [kv][d]
constexpr int LVS = 128 + 8;     // Vs [d][kv]   (transposed for B-fragments)
constexpr int LPS = 128 + 8;     // Ps [q][kv]

__global__ __launch_bounds__(256)
void flash_attn_kernel(const _Float16* __restrict__ Q, const _Float16* __restrict__ K,
                       const _Float16* __restrict__ V, float* __restrict__ O,
                       float scale)
{
    extern __shared__ _Float16 smem[];
    _Float16* Qs = smem;
    _Float16* Ks = Qs + 128 * LQS;
    _Float16* Vs = Ks + 128 * LKS;
    _Float16* Ps = Vs + 128 * LVS;

    const int tid  = threadIdx.x;
    const int lane = tid & 31;
    const int wid  = tid >> 5;
    const int b    = blockIdx.z;
    const int h    = blockIdx.y;
    const int s0   = blockIdx.x * 128;
    const size_t base = (size_t)b * SS * DD + (size_t)h * HDIM;

    // Stage Q tile once, pre-scaled by 1/sqrt(HD)
    #pragma unroll
    for (int i = 0; i < 8; ++i) {
        int idx = tid + i * 256;
        int r = idx >> 4, c8 = idx & 15;
        v8h q = *(const v8h*)(Q + base + (size_t)(s0 + r) * DD + c8 * 8);
        #pragma unroll
        for (int j = 0; j < 8; ++j) q[j] = (_Float16)((float)q[j] * scale);
        *(v8h*)(Qs + r * LQS + c8 * 8) = q;
    }

    const v8f vz = {0.f,0.f,0.f,0.f,0.f,0.f,0.f,0.f};
    v8f acc_o[8];
    #pragma unroll
    for (int nt = 0; nt < 8; ++nt) acc_o[nt] = vz;
    float m_i[8], l_i[8];
    #pragma unroll
    for (int r = 0; r < 8; ++r) { m_i[r] = -3.0e38f; l_i[r] = 0.f; }

    for (int skv = 0; skv < SS; skv += 128) {
        __syncthreads();   // prev iteration finished reading Ks/Vs; Qs ready
        // Stage K [kv][d] and V transposed [d][kv]
        #pragma unroll
        for (int i = 0; i < 8; ++i) {
            int idx = tid + i * 256;
            int r = idx >> 4, c8 = idx & 15;
            v8h kv = *(const v8h*)(K + base + (size_t)(skv + r) * DD + c8 * 8);
            *(v8h*)(Ks + r * LKS + c8 * 8) = kv;
            v8h vv = *(const v8h*)(V + base + (size_t)(skv + r) * DD + c8 * 8);
            #pragma unroll
            for (int j = 0; j < 8; ++j) Vs[(c8 * 8 + j) * LVS + r] = vv[j];
        }
        __syncthreads();

        // S = Q * K^T : 16 q-rows x 128 kv per wave, K-dim = HD = 128
        v8f accs[8];
        #pragma unroll
        for (int nt = 0; nt < 8; ++nt) accs[nt] = vz;
        #pragma unroll
        for (int kk = 0; kk < 4; ++kk) {
            int rowA = wid * 16 + (lane & 15);
            int kbA  = kk * 32 + (lane >> 4) * 8;
            union { v16h v; v8h h[2]; } ua;
            ua.h[0] = *(const v8h*)(Qs + rowA * LQS + kbA);
            ua.h[1] = *(const v8h*)(Qs + rowA * LQS + kbA + 16);
            int kbB = kk * 32 + (lane >> 4) * 8;
            #pragma unroll
            for (int nt = 0; nt < 8; ++nt) {
                int col = nt * 16 + (lane & 15);
                union { v16h v; v8h h[2]; } ub;
                ub.h[0] = *(const v8h*)(Ks + col * LKS + kbB);
                ub.h[1] = *(const v8h*)(Ks + col * LKS + kbB + 16);
                accs[nt] = wmma16x16x32(ua.v, ub.v, accs[nt]);
            }
        }

        // Online softmax (row r of this lane = wid*16 + r + (lane>>4)*8)
        float mnew[8], alpha[8], rs[8];
        #pragma unroll
        for (int r = 0; r < 8; ++r) {
            float mx = m_i[r];
            #pragma unroll
            for (int nt = 0; nt < 8; ++nt) mx = fmaxf(mx, accs[nt][r]);
            #pragma unroll
            for (int off = 1; off < 16; off <<= 1) mx = fmaxf(mx, __shfl_xor(mx, off));
            mnew[r]  = mx;
            alpha[r] = __expf(m_i[r] - mx);
            rs[r]    = 0.f;
        }
        #pragma unroll
        for (int nt = 0; nt < 8; ++nt)
            #pragma unroll
            for (int r = 0; r < 8; ++r) {
                float p = __expf(accs[nt][r] - mnew[r]);
                accs[nt][r] = p;
                rs[r] += p;
            }
        #pragma unroll
        for (int r = 0; r < 8; ++r) {
            #pragma unroll
            for (int off = 1; off < 16; off <<= 1) rs[r] += __shfl_xor(rs[r], off);
            l_i[r] = l_i[r] * alpha[r] + rs[r];
            m_i[r] = mnew[r];
        }
        #pragma unroll
        for (int nt = 0; nt < 8; ++nt)
            #pragma unroll
            for (int r = 0; r < 8; ++r) acc_o[nt][r] *= alpha[r];

        // P (C/D layout) -> LDS f16 -> A-fragment layout. Same-wave LDS is
        // in-order, and each wave only reads its own 16 rows: no barrier.
        #pragma unroll
        for (int nt = 0; nt < 8; ++nt)
            #pragma unroll
            for (int r = 0; r < 8; ++r)
                Ps[(wid * 16 + r + (lane >> 4) * 8) * LPS + nt * 16 + (lane & 15)] =
                    (_Float16)accs[nt][r];

        // O += P @ V  (K-dim = kv tile of 128)
        #pragma unroll
        for (int kk = 0; kk < 4; ++kk) {
            int rowA = wid * 16 + (lane & 15);
            int kbA  = kk * 32 + (lane >> 4) * 8;
            union { v16h v; v8h h[2]; } ua;
            ua.h[0] = *(const v8h*)(Ps + rowA * LPS + kbA);
            ua.h[1] = *(const v8h*)(Ps + rowA * LPS + kbA + 16);
            int kbB = kk * 32 + (lane >> 4) * 8;
            #pragma unroll
            for (int nt = 0; nt < 8; ++nt) {
                int col = nt * 16 + (lane & 15);
                union { v16h v; v8h h[2]; } ub;
                ub.h[0] = *(const v8h*)(Vs + col * LVS + kbB);
                ub.h[1] = *(const v8h*)(Vs + col * LVS + kbB + 16);
                acc_o[nt] = wmma16x16x32(ua.v, ub.v, acc_o[nt]);
            }
        }
    }

    // Normalize and write O (b, s, h*HD + d) in f32
    #pragma unroll
    for (int r = 0; r < 8; ++r) l_i[r] = 1.0f / l_i[r];
    #pragma unroll
    for (int nt = 0; nt < 8; ++nt)
        #pragma unroll
        for (int r = 0; r < 8; ++r) {
            int row = s0 + wid * 16 + r + (lane >> 4) * 8;
            int col = nt * 16 + (lane & 15);
            O[((size_t)b * SS + row) * DD + h * HDIM + col] = acc_o[nt][r] * l_i[r];
        }
}

// ---------------------------------------------------------------------------
extern "C" void kernel_launch(void* const* d_in, const int* in_sizes, int n_in,
                              void* d_out, int out_size, void* d_ws, size_t ws_size,
                              hipStream_t stream)
{
    (void)in_sizes; (void)n_in; (void)out_size; (void)ws_size;
    const float* X  = (const float*)d_in[0];
    const float* Wq = (const float*)d_in[1];
    const float* bq = (const float*)d_in[2];
    const float* Wk = (const float*)d_in[3];
    const float* bk = (const float*)d_in[4];
    const float* Wv = (const float*)d_in[5];
    const float* bv = (const float*)d_in[6];
    const float* Wo = (const float*)d_in[7];
    const float* bo = (const float*)d_in[8];
    float* out = (float*)d_out;

    const int M = MTOT, N = DD, Kd = DD;

    // Workspace: Q/K/V f16 (16MB each) + attention output f32 (32MB)
    _Float16* Qh = (_Float16*)d_ws;
    _Float16* Kh = Qh + (size_t)M * N;
    _Float16* Vh = Kh + (size_t)M * N;
    float*    AO = (float*)(Vh + (size_t)M * N);

    dim3 gg(N / BN, M / BM);   // (8, 64)
    gemm_bias_kernel<true><<<gg, 256, 0, stream>>>(X, Wq, bq, nullptr, Qh, M, N, Kd);
    gemm_bias_kernel<true><<<gg, 256, 0, stream>>>(X, Wk, bk, nullptr, Kh, M, N, Kd);
    gemm_bias_kernel<true><<<gg, 256, 0, stream>>>(X, Wv, bv, nullptr, Vh, M, N, Kd);

    size_t smem = (size_t)(128 * LQS + 128 * LKS + 128 * LVS + 128 * LPS) * sizeof(_Float16);
    flash_attn_kernel<<<dim3(SS / 128, HH, BB), 256, smem, stream>>>(
        Qh, Kh, Vh, AO, 0.08838834764831845f /* 1/sqrt(128) */);

    gemm_bias_kernel<false><<<gg, 256, 0, stream>>>(AO, Wo, bo, out, nullptr, M, N, Kd);
}
